// MoEBlock_9053791060076
// MI455X (gfx1250) — matrix operand order repaired and checked by reading, not compiled, not run on hardware
//
#include <hip/hip_runtime.h>

#define C_DIM 256
#define E_DIM 4
#define SPB   16384          // T*H*W per batch
#define MT    128            // positions per workgroup
#define CP    264            // padded LDS row stride in bf16 elems (rows 16B-aligned)

typedef __attribute__((ext_vector_type(16))) __bf16 bf16x16;
typedef __attribute__((ext_vector_type(8)))  __bf16 bf16x8;
typedef __attribute__((ext_vector_type(8)))  float  f32x8;

// fast reciprocal: single v_rcp_f32 (plenty of accuracy ahead of a bf16 round)
__device__ __forceinline__ float fast_rcp(float x) {
  return __builtin_amdgcn_rcpf(x);
}

// ---------------- weight conversion: fp32 -> bf16 into workspace ----------------
__global__ void moe_convert_weights(const float* __restrict__ w1,
                                    const float* __restrict__ w2,
                                    __bf16* __restrict__ w1b,
                                    __bf16* __restrict__ w2b) {
  const int N1 = E_DIM * C_DIM * C_DIM;   // 262144 elements each
  int idx = blockIdx.x * blockDim.x + threadIdx.x;
  if (idx < N1)          w1b[idx]      = (__bf16)w1[idx];
  else if (idx < 2 * N1) w2b[idx - N1] = (__bf16)w2[idx - N1];
}

// ---------------- fused MoE kernel ----------------
__global__ __launch_bounds__(256, 1)
void moe_fused(const float* __restrict__ x,
               const float* __restrict__ gate_w,
               const float* __restrict__ gate_b,
               const float* __restrict__ b1,
               const float* __restrict__ b2,
               const __bf16* __restrict__ w1b,
               const __bf16* __restrict__ w2b,
               float* __restrict__ out) {
  __shared__ __bf16 xs[MT * CP];          // x tile, bf16 [m][c]
  __shared__ __bf16 hs[MT * CP];          // per-expert hidden tile, bf16 [m][c]
  __shared__ float  wts[MT * E_DIM];      // gating weights per row

  const int tid  = threadIdx.x;
  const int lane = tid & 31;
  const int wave = tid >> 5;
  const int lnN  = lane & 15;             // column within 16-wide N tile
  const int hi   = lane >> 4;             // lane-group (K/M split per WMMA layout)
  const int hi8  = hi * 8;
  const int m0   = wave * 16;             // wave's M-subtile base row

  const int b  = blockIdx.x >> 7;                 // batch (128 blocks per batch)
  const int s0 = (blockIdx.x & 127) * MT;         // spatial base
  const float* xb = x   + (size_t)b * C_DIM * SPB + s0;
  float*       ob = out + (size_t)b * C_DIM * SPB + s0;

  // ---- stage x tile to LDS (fp32 -> bf16), positions contiguous in memory ----
  for (int i = tid; i < MT * C_DIM / 4; i += 256) {
    int m4 = (i & 31) * 4;                // MT/4 == 32
    int c  = i >> 5;
    float4 v = *(const float4*)(xb + (size_t)c * SPB + m4);
    xs[(m4 + 0) * CP + c] = (__bf16)v.x;
    xs[(m4 + 1) * CP + c] = (__bf16)v.y;
    xs[(m4 + 2) * CP + c] = (__bf16)v.z;
    xs[(m4 + 3) * CP + c] = (__bf16)v.w;
  }
  __syncthreads();

  // ---- gating: logits, top-2, softmax -> wts ----
  if (tid < MT) {
    float lg[E_DIM];
    #pragma unroll
    for (int e = 0; e < E_DIM; ++e) lg[e] = gate_b[e];
    for (int c = 0; c < C_DIM; ++c) {
      float xv = (float)xs[tid * CP + c];
      #pragma unroll
      for (int e = 0; e < E_DIM; ++e) lg[e] += xv * gate_w[e * C_DIM + c];
    }
    int i0 = 0;
    #pragma unroll
    for (int e = 1; e < E_DIM; ++e) if (lg[e] > lg[i0]) i0 = e;
    int i1 = (i0 == 0) ? 1 : 0;
    #pragma unroll
    for (int e = 0; e < E_DIM; ++e) if (e != i0 && lg[e] > lg[i1]) i1 = e;
    float e1  = __expf(lg[i1] - lg[i0]);
    float inv = fast_rcp(1.0f + e1);
    #pragma unroll
    for (int e = 0; e < E_DIM; ++e) wts[tid * E_DIM + e] = 0.0f;
    wts[tid * E_DIM + i0] = inv;
    wts[tid * E_DIM + i1] = e1 * inv;
  }
  __syncthreads();

  // ---- register accumulators (D-layout), initialized with residual x (fp32) ----
  f32x8 acc[16];
  #pragma unroll
  for (int nt = 0; nt < 16; ++nt) {
    const float* rp = xb + (size_t)(nt * 16 + lnN) * SPB + m0 + hi8;
    float4 v0 = *(const float4*)rp;
    float4 v1 = *(const float4*)(rp + 4);
    acc[nt][0] = v0.x; acc[nt][1] = v0.y; acc[nt][2] = v0.z; acc[nt][3] = v0.w;
    acc[nt][4] = v1.x; acc[nt][5] = v1.y; acc[nt][6] = v1.z; acc[nt][7] = v1.w;
  }

  // ---- expert loop ----
  for (int e = 0; e < E_DIM; ++e) {
    // layer1: H_e = silu(X @ W1_e^T + b1_e) -> hs (bf16)
    for (int nt = 0; nt < 16; ++nt) {
      f32x8 psum = {};
      const int ch = nt * 16 + lnN;                         // H channel (0..255)
      const __bf16* wrow = w1b + ((size_t)(e * C_DIM + ch)) * C_DIM;
      #pragma unroll
      for (int k0 = 0; k0 < C_DIM; k0 += 32) {
        const __bf16* ap = &xs[(m0 + lnN) * CP + k0 + hi * 16];
        bf16x8 alo = *(const bf16x8*)ap;
        bf16x8 ahi = *(const bf16x8*)(ap + 8);
        bf16x16 a;
        #pragma unroll
        for (int j = 0; j < 8; ++j) { a[j] = alo[j]; a[j + 8] = ahi[j]; }
        bf16x16 bm = *(const bf16x16*)(wrow + k0 + hi * 16);
        psum = __builtin_amdgcn_wmma_f32_16x16x32_bf16(false, a, false, bm,
                                                       (short)0, psum, false, false);
      }
      float bias = b1[e * C_DIM + ch];
      #pragma unroll
      for (int r = 0; r < 8; ++r) {
        float v = psum[r] + bias;
        float s = v * fast_rcp(1.0f + __expf(-v));          // SiLU via v_rcp_f32
        hs[(m0 + hi8 + r) * CP + ch] = (__bf16)s;
      }
    }
    __syncthreads();

    // per-row gating weight for this expert (rows owned by this lane in D layout)
    float wl[8];
    #pragma unroll
    for (int r = 0; r < 8; ++r) wl[r] = wts[(m0 + hi8 + r) * E_DIM + e];

    // layer2: O_e = H_e @ W2_e^T + b2_e ; acc += w[m,e] * O_e
    #pragma unroll
    for (int nt = 0; nt < 16; ++nt) {
      f32x8 psum = {};
      const int co = nt * 16 + lnN;                         // output channel
      const __bf16* wrow = w2b + (size_t)e * C_DIM * C_DIM + (size_t)co * C_DIM;
      #pragma unroll
      for (int k0 = 0; k0 < C_DIM; k0 += 32) {
        const __bf16* ap = &hs[(m0 + lnN) * CP + k0 + hi * 16];
        bf16x8 alo = *(const bf16x8*)ap;
        bf16x8 ahi = *(const bf16x8*)(ap + 8);
        bf16x16 a;
        #pragma unroll
        for (int j = 0; j < 8; ++j) { a[j] = alo[j]; a[j + 8] = ahi[j]; }
        bf16x16 bm = *(const bf16x16*)(wrow + k0 + hi * 16);
        psum = __builtin_amdgcn_wmma_f32_16x16x32_bf16(false, a, false, bm,
                                                       (short)0, psum, false, false);
      }
      float bias = b2[e * C_DIM + co];
      #pragma unroll
      for (int r = 0; r < 8; ++r)
        acc[nt][r] += wl[r] * (psum[r] + bias);
    }
    __syncthreads();
  }

  // ---- store: contiguous along spatial axis (fp32) ----
  #pragma unroll
  for (int nt = 0; nt < 16; ++nt) {
    float* op = ob + (size_t)(nt * 16 + lnN) * SPB + m0 + hi8;
    float4 v0 = make_float4(acc[nt][0], acc[nt][1], acc[nt][2], acc[nt][3]);
    float4 v1 = make_float4(acc[nt][4], acc[nt][5], acc[nt][6], acc[nt][7]);
    *(float4*)op       = v0;
    *(float4*)(op + 4) = v1;
  }
}

extern "C" void kernel_launch(void* const* d_in, const int* in_sizes, int n_in,
                              void* d_out, int out_size, void* d_ws, size_t ws_size,
                              hipStream_t stream) {
  const float* x      = (const float*)d_in[0];
  const float* gate_w = (const float*)d_in[1];
  const float* gate_b = (const float*)d_in[2];
  const float* w1     = (const float*)d_in[3];
  const float* b1     = (const float*)d_in[4];
  const float* w2     = (const float*)d_in[5];
  const float* b2     = (const float*)d_in[6];
  float* out          = (float*)d_out;

  __bf16* w1b = (__bf16*)d_ws;                       // 512 KB
  __bf16* w2b = w1b + E_DIM * C_DIM * C_DIM;         // 512 KB

  // 2 * 262144 elements, 1 per thread
  moe_convert_weights<<<2048, 256, 0, stream>>>(w1, w2, w1b, w2b);

  // 32768 positions / 128 per block = 256 blocks
  moe_fused<<<256, 256, 0, stream>>>(x, gate_w, gate_b, b1, b2, w1b, w2b, out);
}
